// AppearanceDecoder_80900003988268
// MI455X (gfx1250) — matrix operand order repaired and checked by reading, not compiled, not run on hardware
//
#include <hip/hip_runtime.h>
#include <hip/hip_bf16.h>

typedef __bf16 bf16;
typedef __attribute__((ext_vector_type(16))) __bf16 v16bf;
typedef __attribute__((ext_vector_type(8)))  __bf16 v8bf;
typedef __attribute__((ext_vector_type(8)))  float  v8f;

#define EPS 1e-5f

// ---------------- elementwise helpers ----------------
__global__ void k_f32_to_bf16(const float* __restrict__ src, bf16* __restrict__ dst, long n) {
  long i = (long)blockIdx.x * blockDim.x + threadIdx.x;
  if (i < n) dst[i] = (bf16)src[i];
}

__global__ void k_zero_bf16(bf16* __restrict__ dst, long n) {
  long i = (long)blockIdx.x * blockDim.x + threadIdx.x;
  if (i < n) dst[i] = (bf16)0.0f;
}

// 3x3 weights (O,I,ky,kx) f32 -> (O, tap, I) bf16 so GEMM k = tap*256 + ci
__global__ void k_w3x3_pack(const float* __restrict__ src, bf16* __restrict__ dst, long total) {
  long i = (long)blockIdx.x * blockDim.x + threadIdx.x;
  if (i >= total) return;
  int ci = (int)(i & 255);
  long t = i >> 8;                 // o*9 + tap
  int r = (int)(t % 9);
  long o = t / 9;
  dst[i] = (bf16)src[(o * 256 + ci) * 9 + r];
}

// tiled transpose + convert: src (R x Cc) f32 -> dst (Cc x R) bf16, per batch z
__global__ void k_f32_to_bf16_T(const float* __restrict__ src, bf16* __restrict__ dst,
                                int R, int Cc, long sS, long sD) {
  __shared__ bf16 tile[32][33];
  const float* s = src + (long)blockIdx.z * sS;
  bf16* d = dst + (long)blockIdx.z * sD;
  int r0 = blockIdx.y * 32, c0 = blockIdx.x * 32;
  int tx = threadIdx.x & 31, ty = threadIdx.x >> 5;     // 32 x 8
#pragma unroll
  for (int j = 0; j < 32; j += 8)
    tile[ty + j][tx] = (bf16)s[(long)(r0 + ty + j) * Cc + c0 + tx];
  __syncthreads();
#pragma unroll
  for (int j = 0; j < 32; j += 8)
    d[(long)(c0 + ty + j) * R + r0 + tx] = tile[tx][ty + j];
}

// tiled transpose: src (R x Cc) bf16 -> dst (Cc x R) bf16, per batch z
__global__ void k_bf16_T(const bf16* __restrict__ src, bf16* __restrict__ dst,
                         int R, int Cc, long sS, long sD) {
  __shared__ bf16 tile[32][33];
  const bf16* s = src + (long)blockIdx.z * sS;
  bf16* d = dst + (long)blockIdx.z * sD;
  int r0 = blockIdx.y * 32, c0 = blockIdx.x * 32;
  int tx = threadIdx.x & 31, ty = threadIdx.x >> 5;
#pragma unroll
  for (int j = 0; j < 32; j += 8)
    tile[ty + j][tx] = s[(long)(r0 + ty + j) * Cc + c0 + tx];
  __syncthreads();
#pragma unroll
  for (int j = 0; j < 32; j += 8)
    d[(long)(c0 + ty + j) * R + r0 + tx] = tile[tx][ty + j];
}

// pred_embds (B,C,T,Q) -> (B*T, QP, C) bf16, zero-padded rows q>=Q
__global__ void k_pred_pack(const float* __restrict__ pred, bf16* __restrict__ dst,
                            int B, int T, int Q, int C, int QP) {
  long total = (long)B * T * QP * C;
  long i = (long)blockIdx.x * blockDim.x + threadIdx.x;
  if (i >= total) return;
  int c = (int)(i % C);
  long r = i / C;
  int q  = (int)(r % QP);
  int bt = (int)(r / QP);
  int b = bt / T, t = bt % T;
  dst[i] = (q < Q) ? (bf16)pred[(((long)b * C + c) * T + t) * Q + q] : (bf16)0.0f;
}

// ---------------- GroupNorm ----------------
__global__ void k_gn_stats(const float* __restrict__ x, int HW, float* __restrict__ stats) {
  int blk = blockIdx.x;                    // n*32 + g
  const float* p = x + (long)blk * 8 * HW;
  long cnt = 8L * HW;
  float s = 0.f, ss = 0.f;
  for (long i = threadIdx.x; i < cnt; i += blockDim.x) { float v = p[i]; s += v; ss += v * v; }
  __shared__ float rs[256], rq[256];
  rs[threadIdx.x] = s; rq[threadIdx.x] = ss;
  __syncthreads();
  for (int st = 128; st > 0; st >>= 1) {
    if ((int)threadIdx.x < st) { rs[threadIdx.x] += rs[threadIdx.x + st]; rq[threadIdx.x] += rq[threadIdx.x + st]; }
    __syncthreads();
  }
  if (threadIdx.x == 0) {
    float m = rs[0] / (float)cnt;
    float var = rq[0] / (float)cnt - m * m;
    stats[2 * blk]     = m;
    stats[2 * blk + 1] = rsqrtf(var + EPS);
  }
}

__global__ void k_gn_norm(const float* __restrict__ x, const float* __restrict__ stats,
                          const float* __restrict__ gs, const float* __restrict__ gb,
                          bf16* __restrict__ dst, int HW, long total) {
  long i = (long)blockIdx.x * blockDim.x + threadIdx.x;
  if (i >= total) return;
  long chw = i / HW;                       // n*256 + c
  int  c   = (int)(chw % 256);
  long ng  = chw >> 3;                     // n*32 + g
  float m = stats[2 * ng], r = stats[2 * ng + 1];
  dst[i] = (bf16)((x[i] - m) * r * gs[c] + gb[c]);
}

// ---------------- softmax over last dim (one block per row) ----------------
__global__ void k_softmax(const float* __restrict__ s, bf16* __restrict__ a, int D) {
  const float* row = s + (long)blockIdx.x * D;
  bf16* arow = a + (long)blockIdx.x * D;
  __shared__ float red[256];
  float mx = -3.4e38f;
  for (int i = threadIdx.x; i < D; i += 256) mx = fmaxf(mx, row[i]);
  red[threadIdx.x] = mx; __syncthreads();
  for (int st = 128; st > 0; st >>= 1) {
    if ((int)threadIdx.x < st) red[threadIdx.x] = fmaxf(red[threadIdx.x], red[threadIdx.x + st]);
    __syncthreads();
  }
  mx = red[0]; __syncthreads();
  float sum = 0.f;
  for (int i = threadIdx.x; i < D; i += 256) sum += __expf(row[i] - mx);
  red[threadIdx.x] = sum; __syncthreads();
  for (int st = 128; st > 0; st >>= 1) {
    if ((int)threadIdx.x < st) red[threadIdx.x] += red[threadIdx.x + st];
    __syncthreads();
  }
  float inv = 1.f / red[0];
  for (int i = threadIdx.x; i < D; i += 256) arow[i] = (bf16)(__expf(row[i] - mx) * inv);
}

// ---------------- generic WMMA bf16 GEMM ----------------
// C[z] = A[z](MxK) * B[z](KxN) (+bias)(+relu). BM=64 BN=128 BK=32; 8 waves in a
// 4(M)x2(N) grid; each wave computes 16x64 via 4 accumulators. Software
// pipelined: next k-tile is fetched into registers while WMMAs run.
// BMODE 2: B K-contiguous (element B[n*ldb + k]).
// BMODE 1: implicit im2col, 3x3 pad 1, k = tap*256 + ci, image (256, 2^WSH, 2^WSH).
//          A tile must be packed to the same k-order (k_w3x3_pack).
struct GemmParams {
  const bf16* A; long sA; int lda;
  const bf16* B; long sB; int ldb;
  const float* bias;
  float* outF; bf16* outB; long sC; int ldc;
  int M, N, K, rowmax, relu;
};

__device__ __forceinline__ v16bf frag2(const bf16* p0, const bf16* p1) {
  v8bf lo = *(const v8bf*)p0;
  v8bf hi = *(const v8bf*)p1;
  return __builtin_shufflevector(lo, hi, 0,1,2,3,4,5,6,7,8,9,10,11,12,13,14,15);
}

template<int BMODE, int WSH>
__launch_bounds__(256)
__global__ void k_gemm(GemmParams p) {
  __shared__ bf16 As[64][40];              // row-major MxK tile, stride 80B
  __shared__ bf16 Bt[128][40];             // transposed B tile: Bt[n][k], stride 80B
  const int W = 1 << WSH, HWsz = 1 << (2 * WSH);
  const bf16* A = p.A + (long)blockIdx.z * p.sA;
  const bf16* B = p.B + (long)blockIdx.z * p.sB;
  int m0 = blockIdx.y * 64, n0 = blockIdx.x * 128;
  int tid = threadIdx.x, lane = tid & 31, wave = tid >> 5;
  int wm = wave & 3, wn = wave >> 2;
  int ln = lane & 15;
  int ml  = wm * 16 + ln;
  int kbA = (lane >> 4) << 4;
  int kbB = (lane >> 4) << 4;
  int nbase = wn * 64 + ln;
  v8f acc[4] = {{}, {}, {}, {}};

  // staging coordinates
  int ar = tid >> 2, acs = (tid & 3) << 3;            // A: 64 rows x 4 segs
  int bn = tid >> 1, bks = (tid & 1) << 4;            // B: 128 n-rows x 2 k-segs
  const bf16* pA = A + (long)(m0 + ar) * p.lda + acs;
  const bf16* pB = B + (long)(n0 + bn) * p.ldb + bks; // BMODE==2 only
  int iy = 0, ix = 0;
  if (BMODE == 1) { int ng = n0 + bn; iy = ng >> WSH; ix = ng & (W - 1); }

  uint4 ra, rb0, rb1;
  auto load_tile = [&](int k0) {
    ra = *(const uint4*)(pA + k0);
    if (BMODE == 2) {
      rb0 = *(const uint4*)(pB + k0);
      rb1 = *(const uint4*)(pB + k0 + 8);
    } else {
      // one 3x3 tap per 32-wide k tile (32 | 256); tap uniform across tile
      int tap = k0 >> 8;                   // 0..8
      int ky = tap / 3 - 1, kx = tap % 3 - 1;   // scalar
      int sy = iy + ky, sx = ix + kx;
      int ci0 = (k0 & 255) + bks;
      if (sy >= 0 && sy < W && sx >= 0 && sx < W) {
        const bf16* src = B + (long)ci0 * HWsz + (sy << WSH) + sx;
        bf16 tmp[16];
#pragma unroll
        for (int j = 0; j < 16; ++j) tmp[j] = src[(long)j * HWsz];  // immediate offsets
        rb0 = *(uint4*)&tmp[0];
        rb1 = *(uint4*)&tmp[8];
      } else {
        uint4 z = {0u, 0u, 0u, 0u};
        rb0 = z; rb1 = z;
      }
    }
  };

  load_tile(0);
  for (int k0 = 0; k0 < p.K; k0 += 32) {
    // commit staged registers to LDS
    *(uint4*)(&As[ar][acs])     = ra;
    *(uint4*)(&Bt[bn][bks])     = rb0;
    *(uint4*)(&Bt[bn][bks + 8]) = rb1;
    __syncthreads();
    if (k0 + 32 < p.K) load_tile(k0 + 32);  // overlap next fetch with WMMAs

    v16bf af = frag2(&As[ml][kbA], &As[ml][kbA + 8]);
#pragma unroll
    for (int t = 0; t < 4; ++t) {
      const bf16* bp = &Bt[nbase + t * 16][kbB];
      v16bf bfrag = frag2(bp, bp + 8);
      acc[t] = __builtin_amdgcn_wmma_f32_16x16x32_bf16(false, af, false, bfrag,
                                                       (short)0, acc[t], false, false);
    }
    __syncthreads();
  }

  // epilogue: bias + relu, f32 or bf16 output, row clipping
  int rbase = m0 + wm * 16 + ((lane >> 4) << 3);
#pragma unroll
  for (int t = 0; t < 4; ++t) {
    int c = n0 + wn * 64 + t * 16 + ln;
    float bv = p.bias ? p.bias[c] : 0.f;
#pragma unroll
    for (int j = 0; j < 8; ++j) {
      int r = rbase + j;
      if (r < p.rowmax) {
        float v = acc[t][j] + bv;
        if (p.relu) v = fmaxf(v, 0.f);
        long o = (long)blockIdx.z * p.sC + (long)r * p.ldc + c;
        if (p.outF) p.outF[o] = v;
        else        p.outB[o] = (bf16)v;
      }
    }
  }
}

static void launch_gemm(hipStream_t st, const GemmParams& p, int bmode, int wsh, int batch) {
  dim3 g((unsigned)(p.N / 128), (unsigned)((p.M + 63) / 64), (unsigned)batch);
  if (bmode == 2)      k_gemm<2, 0><<<g, 256, 0, st>>>(p);
  else if (wsh == 7)   k_gemm<1, 7><<<g, 256, 0, st>>>(p);
  else if (wsh == 6)   k_gemm<1, 6><<<g, 256, 0, st>>>(p);
  else                 k_gemm<1, 5><<<g, 256, 0, st>>>(p);
}

// ---------------- host orchestration ----------------
extern "C" void kernel_launch(void* const* d_in, const int* in_sizes, int n_in,
                              void* d_out, int out_size, void* d_ws, size_t ws_size,
                              hipStream_t stream) {
  (void)in_sizes; (void)n_in; (void)out_size; (void)ws_size;
  const int B = 2, T = 2, Q = 100, C = 256, QP = 128, MP = 448;
  const int Cin[3] = {256, 512, 1024};
  const int HWs[3] = {128, 64, 32};
  const int Wsh[3] = {7, 6, 5};

  const float* pred = (const float*)d_in[0];
  const float* feat[3] = {(const float*)d_in[1], (const float*)d_in[2], (const float*)d_in[3]};
  const float* aggp[4]; const float* embp[4]; const float* trkp[4];
  for (int j = 0; j < 4; ++j) { aggp[j] = (const float*)d_in[28 + j];
                                embp[j] = (const float*)d_in[32 + j];
                                trkp[j] = (const float*)d_in[36 + j]; }

  char* ws = (char*)d_ws;
  size_t off = 0;
  auto alloc = [&](size_t bytes) -> void* {
    void* p = ws + off; off += (bytes + 255) & ~(size_t)255; return p;
  };
  float* convOut = (float*)alloc(4ul * 256 * 16384 * 4);
  bf16*  bfA     = (bf16*) alloc(4ul * 256 * 16384 * 2);   // normalized f (channel-major)
  bf16*  bfB     = (bf16*) alloc(4ul * 256 * 16384 * 2);   // normalized conv1 out (image for im2col)
  bf16*  bfT     = (bf16*) alloc(4ul * 256 * 16384 * 2);   // transposed feat, later transposed f
  float* scores  = (float*)alloc(4ul * QP * 16384 * 4);
  bf16*  attn    = (bf16*) alloc(4ul * QP * 16384 * 2);
  float* stats   = (float*)alloc(4ul * 32 * 2 * 4);
  bf16*  predbf  = (bf16*) alloc(4ul * QP * 256 * 2);
  float* qcat    = (float*)alloc(400ul * 768 * 4);
  bf16*  X0      = (bf16*) alloc((size_t)MP * 768 * 2);
  bf16*  Hb1     = (bf16*) alloc((size_t)MP * 256 * 2);
  bf16*  Hb2     = (bf16*) alloc((size_t)MP * 256 * 2);
  bf16*  wArena  = (bf16*) alloc(4ul * 1024 * 1024 * 2);

  long woff = 0;
  auto wslot = [&](long n) -> bf16* { bf16* p = wArena + woff; woff += (n + 127) & ~127L; return p; };
  bf16* w1bf[3]; bf16* w2bf[3];
  for (int l = 0; l < 3; ++l) { w1bf[l] = wslot(256L * Cin[l]); w2bf[l] = wslot(256L * 256 * 9); }
  bf16* aw1 = wslot(256L * 768); bf16* aw2 = wslot(256L * 256);
  bf16* ew1 = wslot(256L * 256); bf16* ew2 = wslot(256L * 256);
  bf16* tw1 = wslot(256L * 256); bf16* tw2 = wslot(256L * 256);

  auto cvt = [&](const float* s, bf16* d, long n) {
    k_f32_to_bf16<<<dim3((unsigned)((n + 255) / 256)), 256, 0, stream>>>(s, d, n);
  };

  for (int l = 0; l < 3; ++l) {
    cvt((const float*)d_in[4 + 8 * l + 0], w1bf[l], 256L * Cin[l]);
    { long tot = 256L * 2304;             // (O,I,3,3) -> (O, tap, I) bf16
      k_w3x3_pack<<<dim3((unsigned)((tot + 255) / 256)), 256, 0, stream>>>(
          (const float*)d_in[4 + 8 * l + 4], w2bf[l], tot); }
  }
  cvt(aggp[0], aw1, 256L * 768); cvt(aggp[2], aw2, 256L * 256);
  cvt(embp[0], ew1, 256L * 256); cvt(embp[2], ew2, 256L * 256);
  cvt(trkp[0], tw1, 256L * 256); cvt(trkp[2], tw2, 256L * 256);

  { long tot = 4L * QP * 256;
    k_pred_pack<<<dim3((unsigned)((tot + 255) / 256)), 256, 0, stream>>>(pred, predbf, B, T, Q, C, QP); }

  for (int l = 0; l < 3; ++l) {
    int D = HWs[l] * HWs[l];
    const float* b1  = (const float*)d_in[4 + 8 * l + 1];
    const float* gs1 = (const float*)d_in[4 + 8 * l + 2];
    const float* gb1 = (const float*)d_in[4 + 8 * l + 3];
    const float* b2  = (const float*)d_in[4 + 8 * l + 5];
    const float* gs2 = (const float*)d_in[4 + 8 * l + 6];
    const float* gb2 = (const float*)d_in[4 + 8 * l + 7];

    // feat (Cin,D) f32 -> bfT (D,Cin) bf16 (transposed, so GEMM B is K-contiguous)
    { dim3 g((unsigned)(D / 32), (unsigned)(Cin[l] / 32), 4);
      k_f32_to_bf16_T<<<g, 256, 0, stream>>>(feat[l], bfT, Cin[l], D,
                                             (long)Cin[l] * D, (long)Cin[l] * D); }

    // 1x1 conv: (256 x Cin) @ (Cin x D)
    GemmParams p1 = {};
    p1.A = w1bf[l]; p1.sA = 0; p1.lda = Cin[l];
    p1.B = bfT; p1.sB = (long)Cin[l] * D; p1.ldb = Cin[l];
    p1.bias = b1; p1.outF = convOut; p1.sC = 256L * D; p1.ldc = D;
    p1.M = 256; p1.N = D; p1.K = Cin[l]; p1.rowmax = 256; p1.relu = 0;
    launch_gemm(stream, p1, 2, 0, 4);

    k_gn_stats<<<4 * 32, 256, 0, stream>>>(convOut, D, stats);
    { long tot = 4L * 256 * D;
      k_gn_norm<<<dim3((unsigned)((tot + 255) / 256)), 256, 0, stream>>>(convOut, stats, gs1, gb1, bfB, D, tot); }

    // 3x3 conv as implicit-im2col GEMM: (256 x 2304) @ (2304 x D), k = tap*256+ci
    GemmParams p2 = p1;
    p2.A = w2bf[l]; p2.lda = 2304; p2.K = 2304;
    p2.B = bfB; p2.sB = 256L * D; p2.ldb = 0;
    p2.bias = b2;
    launch_gemm(stream, p2, 1, Wsh[l], 4);

    k_gn_stats<<<4 * 32, 256, 0, stream>>>(convOut, D, stats);
    { long tot = 4L * 256 * D;
      k_gn_norm<<<dim3((unsigned)((tot + 255) / 256)), 256, 0, stream>>>(convOut, stats, gs2, gb2, bfA, D, tot); }

    // f (256,D) -> fT (D,256) so the score GEMM's B is K-contiguous
    { dim3 g((unsigned)(D / 32), 8, 4);
      k_bf16_T<<<g, 256, 0, stream>>>(bfA, bfT, 256, D, 256L * D, 256L * D); }

    // attention scores: (QP x 256) @ (256 x D) per (b,t)
    GemmParams pa = {};
    pa.A = predbf; pa.sA = (long)QP * 256; pa.lda = 256;
    pa.B = bfT; pa.sB = 256L * D; pa.ldb = 256;
    pa.outF = scores; pa.sC = (long)QP * D; pa.ldc = D;
    pa.M = QP; pa.N = D; pa.K = 256; pa.rowmax = QP; pa.relu = 0;
    launch_gemm(stream, pa, 2, 0, 4);

    k_softmax<<<4 * QP, 256, 0, stream>>>(scores, attn, D);

    // queries: (QP x D) @ f^T (D x 256) into qcat columns [l*256, (l+1)*256)
    GemmParams pq = {};
    pq.A = attn; pq.sA = (long)QP * D; pq.lda = D;
    pq.B = bfA; pq.sB = 256L * D; pq.ldb = D;
    pq.outF = qcat + l * 256; pq.sC = 100L * 768; pq.ldc = 768;
    pq.M = QP; pq.N = 256; pq.K = D; pq.rowmax = 100; pq.relu = 0;
    launch_gemm(stream, pq, 2, 0, 4);
  }

  // MLP chain: rows padded 400 -> 448
  { long tot = (long)MP * 768;
    k_zero_bf16<<<dim3((unsigned)((tot + 255) / 256)), 256, 0, stream>>>(X0, tot); }
  cvt(qcat, X0, 400L * 768);

  auto mlp = [&](const bf16* A, int K, const bf16* Wt, int ldb, const float* bias,
                 bf16* outB, float* outF, int rowmax, int relu) {
    GemmParams p = {};
    p.A = A; p.sA = 0; p.lda = K;
    p.B = Wt; p.sB = 0; p.ldb = ldb;
    p.bias = bias; p.outF = outF; p.outB = outB; p.sC = 0; p.ldc = 256;
    p.M = MP; p.N = 256; p.K = K; p.rowmax = rowmax; p.relu = relu;
    launch_gemm(stream, p, 2, 0, 1);
  };
  mlp(X0,  768, aw1, 768, aggp[1], Hb1, nullptr, MP, 1);
  mlp(Hb1, 256, aw2, 256, aggp[3], Hb2, nullptr, MP, 0);
  mlp(Hb2, 256, ew1, 256, embp[1], Hb1, nullptr, MP, 1);
  mlp(Hb1, 256, ew2, 256, embp[3], Hb2, nullptr, MP, 0);
  mlp(Hb2, 256, tw1, 256, trkp[1], Hb1, nullptr, MP, 1);
  mlp(Hb1, 256, tw2, 256, trkp[3], nullptr, (float*)d_out, 400, 0);
}
